// WindowAttention_66889820668062
// MI455X (gfx1250) — matrix operand order repaired and checked by reading, not compiled, not run on hardware
//
#include <hip/hip_runtime.h>

typedef _Float16 f16;
typedef __attribute__((ext_vector_type(16))) _Float16 v16h;
typedef __attribute__((ext_vector_type(8)))  _Float16 v8h;
typedef __attribute__((ext_vector_type(8)))  float    v8f;

#define T_TOK 49
#define TPAD  64
#define CDIM  128
#define NH    4
#define HD    32
#define NWIN  64
#define QK_SCALE 0.17677669529663687f   // 32^-0.5

#define MSK_LDS_OFF 147456u             // byte offset of msk region in dynamic LDS

union FragU { v16h v; v8h h[2]; };

// A-operand (M x K, 16x32 f16): lane r<16 -> row M=r, K = hi*16+0..15
__device__ __forceinline__ v16h load_a(const f16* base, int stride, int row0, int col0, int lane) {
    const int r = lane & 15, hi = (lane >> 4) & 1;
    return *(const v16h*)(base + (size_t)(row0 + r) * stride + col0 + hi * 16);
}

// B-operand (K x N, 32x16 f16): lane r<16 -> col N=r holds K=0..15; lane r+16 -> K=16..31
// source is N-major rows of length `stride` along K -> one contiguous 32B read per lane
__device__ __forceinline__ v16h load_b(const f16* base, int stride, int row0, int col0, int lane) {
    const int r = lane & 15, hi = (lane >> 4) & 1;
    return *(const v16h*)(base + (size_t)(row0 + r) * stride + col0 + hi * 16);
}

__device__ __forceinline__ v8f wmma16(v16h a, v16h b, v8f c) {
    return __builtin_amdgcn_wmma_f32_16x16x32_f16(false, a, false, b, (short)0, c, false, false);
}

__device__ __forceinline__ v8h pack8(v8f a) {
    v8h r;
    for (int j = 0; j < 8; ++j) r[j] = (f16)a[j];
    return r;
}

// gfx1250 async global->LDS copy (ASYNCcnt-tracked), inline asm per CDNA5 ISA
__device__ __forceinline__ void async_copy_b32(unsigned lds_byte_off, const float* gaddr) {
    asm volatile("global_load_async_to_lds_b32 %0, %1, off"
                 :: "v"(lds_byte_off), "v"(gaddr) : "memory");
}
__device__ __forceinline__ void wait_asynccnt0() {
    asm volatile("s_wait_asynccnt 0x0" ::: "memory");
}

// ---- pre-pass: fp32 -> fp16 weight conversion into workspace ----
__global__ void convert_weights_kernel(const float* __restrict__ wqkv,
                                       const float* __restrict__ wproj,
                                       f16* __restrict__ wqkv_h,
                                       f16* __restrict__ wproj_h) {
    int i = blockIdx.x * blockDim.x + threadIdx.x;
    if (i < 3 * CDIM * CDIM) wqkv_h[i] = (f16)wqkv[i];
    if (i < CDIM * CDIM)     wproj_h[i] = (f16)wproj[i];
}

// ---- fused window attention: one block (4 wave32) per window ----
__global__ __launch_bounds__(128) void win_attn_kernel(
    const float* __restrict__ x, const float* __restrict__ mask,
    const f16* __restrict__ wqkv, const float* __restrict__ bqkv,
    const f16* __restrict__ wproj, const float* __restrict__ bproj,
    float* __restrict__ out)
{
    extern __shared__ char smem[];
    f16*   xs  = (f16*)smem;                          // [64][128] f16 (phase 1, aliases sc)
    float* sc  = (float*)smem;                        // [4][64][64] f32 scores (phase 2+)
    f16*   qs  = (f16*)(smem + 65536);                // [4][64][32]
    f16*   ks  = (f16*)(smem + 65536 + 16384);        // [4][64][32]
    f16*   vts = (f16*)(smem + 65536 + 32768);        // [4][32][64]  (V transposed)
    f16*   pr  = (f16*)(smem + 65536 + 49152);        // [4][64][64] f16 probs
    f16*   ys  = (f16*)(smem + MSK_LDS_OFF);          // [64][128] f16 (phase 4+)
    float* msk = (float*)(smem + MSK_LDS_OFF);        // [49][49] f32, aliases ys (phase 1-3)

    const int tid  = threadIdx.x;
    const int lane = tid & 31;
    const int r    = lane & 15;
    const int hi   = (lane >> 4) & 1;
    // force wave-uniform scalar: kills EXEC-masked branches / moves addressing to SALU
    const int wv   = __builtin_amdgcn_readfirstlane(tid >> 5);   // wave id == head id
    const int b    = blockIdx.x;
    const int wm   = b & (NWIN - 1);   // window-mask index = b % nW

    // ---- phase 1a: async-stage mask window (f32), stage x[b] as f16 (zero-padded) ----
    {
        const float* mg = mask + (size_t)wm * T_TOK * T_TOK;
        for (int i = tid; i < T_TOK * T_TOK; i += 128)
            async_copy_b32(MSK_LDS_OFF + (unsigned)i * 4u, mg + i);
    }
    const float* xg = x + (size_t)b * T_TOK * CDIM;
    for (int i = tid * 4; i < TPAD * CDIM; i += 128 * 4) {
        f16 h0 = (f16)0.f, h1 = (f16)0.f, h2 = (f16)0.f, h3 = (f16)0.f;
        if (i < T_TOK * CDIM) {
            float4 f = *(const float4*)(xg + i);
            h0 = (f16)f.x; h1 = (f16)f.y; h2 = (f16)f.z; h3 = (f16)f.w;
        }
        xs[i] = h0; xs[i + 1] = h1; xs[i + 2] = h2; xs[i + 3] = h3;
    }
    wait_asynccnt0();      // each wave retires its async copies before the barrier
    __syncthreads();

    // ---- phase 1b: qkv^T = Wqkv * x^T  (M=384 ch, N=64 tok, K=128) ----
    for (int mti = 0; mti < 6; ++mti) {
        const int chb = __builtin_amdgcn_readfirstlane((wv * 6 + mti) * 16); // uniform
        const int s   = __builtin_amdgcn_readfirstlane(chb >> 7);            // 0=q 1=k 2=v
        const int h   = __builtin_amdgcn_readfirstlane((chb >> 5) & 3);      // head
        const int d0  = (chb & 31) + hi * 8;       // lane's first head-dim
        float4 bb0 = *(const float4*)(bqkv + chb + hi * 8);
        float4 bb1 = *(const float4*)(bqkv + chb + hi * 8 + 4);
        v16h aw[4];
        for (int kk = 0; kk < 4; ++kk) aw[kk] = load_a(wqkv, CDIM, chb, kk * 32, lane);
        for (int nt = 0; nt < 4; ++nt) {
            v8f acc = {bb0.x, bb0.y, bb0.z, bb0.w, bb1.x, bb1.y, bb1.z, bb1.w};
            for (int kk = 0; kk < 4; ++kk) {
                v16h bx = load_b(xs, CDIM, nt * 16, kk * 32, lane);
                acc = wmma16(aw[kk], bx, acc);
            }
            const int t = nt * 16 + r;      // lane's token
            if (s == 0) {
                *(v8h*)(qs + ((size_t)(h * TPAD + t)) * HD + d0) = pack8(acc);
            } else if (s == 1) {
                *(v8h*)(ks + ((size_t)(h * TPAD + t)) * HD + d0) = pack8(acc);
            } else {
                f16* dst = vts + ((size_t)(h * HD + d0)) * TPAD + t;  // transpose V
                for (int j = 0; j < 8; ++j) dst[j * TPAD] = (f16)acc[j];
            }
        }
    }
    __syncthreads();   // also retires xs before sc overwrites it

    // ---- phase 2: sc[q][k] = scale * (K Q^T)  (M=k_t, N=q_t, K=32) ----
    {
        const f16* qh = qs + (size_t)wv * TPAD * HD;
        const f16* kh = ks + (size_t)wv * TPAD * HD;
        float* sh = sc + (size_t)wv * TPAD * TPAD;
        for (int mt = 0; mt < 4; ++mt) {
            v16h ak = load_a(kh, HD, mt * 16, 0, lane);
            for (int nt = 0; nt < 4; ++nt) {
                v16h bq = load_b(qh, HD, nt * 16, 0, lane);
                v8f acc = {0.f, 0.f, 0.f, 0.f, 0.f, 0.f, 0.f, 0.f};
                acc = wmma16(ak, bq, acc);
                v8f sv;
                for (int j = 0; j < 8; ++j) sv[j] = acc[j] * QK_SCALE;
                const int q = nt * 16 + r;
                *(v8f*)(sh + (size_t)q * TPAD + mt * 16 + hi * 8) = sv;
            }
        }
    }
    // intra-wave only (wave == head): LDS ops in-order per wave, no barrier needed

    // ---- phase 3: +mask, softmax rows, f16 probs; lane owns rows lane, lane+32 ----
    {
        float* sh = sc + (size_t)wv * TPAD * TPAD;
        f16*   ph = pr + (size_t)wv * TPAD * TPAD;
        for (int rr = 0; rr < 2; ++rr) {
            const int t = lane + rr * 32;
            f16* prow = ph + (size_t)t * TPAD;
            if (t < T_TOK) {
                float* srow = sh + (size_t)t * TPAD;
                const float* mrow = msk + (size_t)t * T_TOK;
                float mx = -3.0e38f;
                for (int j = 0; j < T_TOK; ++j) {
                    float val = srow[j] + mrow[j];
                    srow[j] = val;
                    mx = fmaxf(mx, val);
                }
                float sum = 0.f;
                for (int j = 0; j < T_TOK; ++j) {
                    float e = __expf(srow[j] - mx);
                    srow[j] = e;
                    sum += e;
                }
                const float inv = 1.0f / sum;
                for (int j = 0; j < T_TOK; ++j) prow[j] = (f16)(srow[j] * inv);
                for (int j = T_TOK; j < TPAD; ++j) prow[j] = (f16)0.f;   // padded keys
            } else {
                for (int j = 0; j < TPAD; ++j) prow[j] = (f16)0.f;       // padded queries
            }
        }
    }
    __syncthreads();   // msk region is about to be overwritten by ys

    // ---- phase 4: ys[t][head cols] = (V^T P^T)^T  (M=d, N=q_t, K=64) ----
    {
        const f16* ph = pr + (size_t)wv * TPAD * TPAD;
        const f16* vh = vts + (size_t)wv * HD * TPAD;
        for (int mt = 0; mt < 2; ++mt) {
            v16h av[2];
            for (int kk = 0; kk < 2; ++kk) av[kk] = load_a(vh, TPAD, mt * 16, kk * 32, lane);
            for (int nt = 0; nt < 4; ++nt) {
                v8f acc = {0.f, 0.f, 0.f, 0.f, 0.f, 0.f, 0.f, 0.f};
                for (int kk = 0; kk < 2; ++kk) {
                    v16h bp = load_b(ph, TPAD, nt * 16, kk * 32, lane);
                    acc = wmma16(av[kk], bp, acc);
                }
                const int t = nt * 16 + r;
                *(v8h*)(ys + (size_t)t * CDIM + wv * HD + mt * 16 + hi * 8) = pack8(acc);
            }
        }
    }
    __syncthreads();

    // ---- phase 5: out^T = Wproj * ys^T + b  (M=128 ch, N=64 tok, K=128) ----
    {
        float* og = out + (size_t)b * T_TOK * CDIM;
        for (int mti = 0; mti < 2; ++mti) {
            const int chb = __builtin_amdgcn_readfirstlane(wv * 32 + mti * 16); // uniform
            const int ch0 = chb + hi * 8;             // lane's first output channel
            float4 bb0 = *(const float4*)(bproj + ch0);
            float4 bb1 = *(const float4*)(bproj + ch0 + 4);
            v16h aw[4];
            for (int kk = 0; kk < 4; ++kk) aw[kk] = load_a(wproj, CDIM, chb, kk * 32, lane);
            for (int nt = 0; nt < 4; ++nt) {
                v8f acc = {bb0.x, bb0.y, bb0.z, bb0.w, bb1.x, bb1.y, bb1.z, bb1.w};
                for (int kk = 0; kk < 4; ++kk) {
                    v16h by = load_b(ys, CDIM, nt * 16, kk * 32, lane);
                    acc = wmma16(aw[kk], by, acc);
                }
                const int t = nt * 16 + r;
                if (t < T_TOK) {
                    float* p = og + (size_t)t * CDIM + ch0;
                    *(float4*)p       = make_float4(acc[0], acc[1], acc[2], acc[3]);
                    *(float4*)(p + 4) = make_float4(acc[4], acc[5], acc[6], acc[7]);
                }
            }
        }
    }
}

extern "C" void kernel_launch(void* const* d_in, const int* in_sizes, int n_in,
                              void* d_out, int out_size, void* d_ws, size_t ws_size,
                              hipStream_t stream) {
    const float* x     = (const float*)d_in[0];
    const float* mask  = (const float*)d_in[1];
    const float* wqkv  = (const float*)d_in[2];
    const float* bqkv  = (const float*)d_in[3];
    const float* wproj = (const float*)d_in[4];
    const float* bproj = (const float*)d_in[5];
    float* out = (float*)d_out;

    f16* wqkv_h  = (f16*)d_ws;                                          // 98304 B
    f16* wproj_h = (f16*)((char*)d_ws + 3 * CDIM * CDIM * sizeof(f16)); // +32768 B

    convert_weights_kernel<<<(3 * CDIM * CDIM + 255) / 256, 256, 0, stream>>>(
        wqkv, wproj, wqkv_h, wproj_h);

    const int BW = in_sizes[0] / (T_TOK * CDIM);   // 8192
    const size_t shmem = 160 * 1024;               // 2 workgroups / WGP (320 KB)
    win_attn_kernel<<<BW, 128, shmem, stream>>>(x, mask, wqkv_h, bqkv, wproj_h, bproj, out);
    (void)n_in; (void)out_size; (void)ws_size;
}
